// LicitacionGNN_8839042695664
// MI455X (gfx1250) — compile-verified, matrix operand order checked by reading, not run on hardware
//
#include <hip/hip_runtime.h>
#include <hip/hip_bf16.h>

// ---------------------------------------------------------------------------
// LicitacionGNN on MI455X (gfx1250, wave32):
//   phase 1: edge scatter-sum (L2-resident f32 atomics)  -- bandwidth bound
//   phase 2: pack activations/weights into WMMA-native bf16 fragments
//   phase 3: fused stacked-K GEMM via v_wmma_f32_16x16x32_bf16 + bias + relu
// ---------------------------------------------------------------------------

typedef __attribute__((ext_vector_type(16))) __bf16 v16bf;
typedef __attribute__((ext_vector_type(8)))  __bf16 v8bf;
typedef __attribute__((ext_vector_type(8)))  float  v8f;
typedef __attribute__((ext_vector_type(8)))  int    v8i;

#define N_LIC 40000
#define D_LIC 256
#define D_EMP 128
#define D_CON 192
#define HDIM  256
#define KA    576   // stacked activation cols: 128 (agg_p) + 256 (x_lic) + 192 (agg_q)
#define KW    832   // stacked weight rows: 128 + 256 + 192 + 256
#define NKT_W 26    // KW/32 weight k-tiles
#define NKT_A 18    // KA/32 activation k-tiles

__device__ __forceinline__ unsigned short f32_to_bf16_bits(float f) {
    union { float f; unsigned u; } x; x.f = f;
    unsigned u = x.u;
    u += 0x7FFFu + ((u >> 16) & 1u);          // round-to-nearest-even
    return (unsigned short)(u >> 16);
}

// ---------------------------------------------------------------------------
__global__ void zero_f32(float* __restrict__ p, long n) {
    long i = (long)blockIdx.x * blockDim.x + threadIdx.x;
    if (i < n) p[i] = 0.0f;
}

// One wave per edge: 32 lanes x float4 = 128 feats; f32 atomics resolve in L2.
__global__ __launch_bounds__(256)
void scatter_p(const float* __restrict__ x_emp,
               const int* __restrict__ src, const int* __restrict__ dst,
               float* __restrict__ agg, float* __restrict__ deg, int n_edges) {
    int wave = (blockIdx.x * blockDim.x + threadIdx.x) >> 5;
    int lane = threadIdx.x & 31;
    if (wave >= n_edges) return;
    int s = src[wave], d = dst[wave];
    const float4 v = *(const float4*)(x_emp + (size_t)s * D_EMP + lane * 4);
    float* ap = agg + (size_t)d * D_EMP + lane * 4;
    __hip_atomic_fetch_add(ap + 0, v.x, __ATOMIC_RELAXED, __HIP_MEMORY_SCOPE_AGENT);
    __hip_atomic_fetch_add(ap + 1, v.y, __ATOMIC_RELAXED, __HIP_MEMORY_SCOPE_AGENT);
    __hip_atomic_fetch_add(ap + 2, v.z, __ATOMIC_RELAXED, __HIP_MEMORY_SCOPE_AGENT);
    __hip_atomic_fetch_add(ap + 3, v.w, __ATOMIC_RELAXED, __HIP_MEMORY_SCOPE_AGENT);
    if (lane == 0)
        __hip_atomic_fetch_add(deg + d, 1.0f, __ATOMIC_RELAXED, __HIP_MEMORY_SCOPE_AGENT);
}

// One wave per edge: 6 stride-32 coalesced floats = 192 feats.
__global__ __launch_bounds__(256)
void scatter_q(const float* __restrict__ x_con,
               const int* __restrict__ src, const int* __restrict__ dst,
               float* __restrict__ agg, float* __restrict__ deg, int n_edges) {
    int wave = (blockIdx.x * blockDim.x + threadIdx.x) >> 5;
    int lane = threadIdx.x & 31;
    if (wave >= n_edges) return;
    int s = src[wave], d = dst[wave];
    const float* xr = x_con + (size_t)s * D_CON;
    float*       ar = agg   + (size_t)d * D_CON;
#pragma unroll
    for (int j = 0; j < 6; ++j) {
        float v = xr[lane + j * 32];
        __hip_atomic_fetch_add(ar + lane + j * 32, v,
                               __ATOMIC_RELAXED, __HIP_MEMORY_SCOPE_AGENT);
    }
    if (lane == 0)
        __hip_atomic_fetch_add(deg + d, 1.0f, __ATOMIC_RELAXED, __HIP_MEMORY_SCOPE_AGENT);
}

// Stacked bf16 activation matrix Acat[40000 x 576] =
//   [agg_p/max(deg_p,1) | x_lic | agg_q/max(deg_q,1)]  (x_lic stored once,
//   reused for both Wr_p and Wr_q products).
__global__ __launch_bounds__(576)
void build_acat(const float* __restrict__ x_lic,
                const float* __restrict__ agg_p, const float* __restrict__ agg_q,
                const float* __restrict__ deg_p, const float* __restrict__ deg_q,
                unsigned short* __restrict__ Acat) {
    const int row = blockIdx.x;
    const int c   = threadIdx.x;
    float v;
    if (c < 128)      v = agg_p[(size_t)row * D_EMP + c] * (1.0f / fmaxf(deg_p[row], 1.0f));
    else if (c < 384) v = x_lic[(size_t)row * D_LIC + (c - 128)];
    else              v = agg_q[(size_t)row * D_CON + (c - 384)] * (1.0f / fmaxf(deg_q[row], 1.0f));
    Acat[(size_t)row * KA + c] = f32_to_bf16_bits(v);
}

__device__ __forceinline__ float wsrc_at(int r, int c,
                                         const float* Wl_p, const float* Wr_p,
                                         const float* Wl_q, const float* Wr_q) {
    if (r < 128) return Wl_p[r * HDIM + c];
    if (r < 384) return Wr_p[(r - 128) * HDIM + c];
    if (r < 576) return Wl_q[(r - 384) * HDIM + c];
    return Wr_q[(r - 576) * HDIM + c];
}

// Pre-swizzle stacked weights into WMMA-B register layout:
//   Wfrag[kt][ntile][lane][v] : dword = bf16x2 of rows (kt*32+klo, kt*32+klo+1),
//   col n = ntile*16 + lane%16, klo = ((v&4)<<2) + 8*(lane/16) + (v&3)*2.
// A B fragment is then one contiguous 32B per-lane load (2 x b128).
__global__ void pack_w_frag(const float* __restrict__ Wl_p, const float* __restrict__ Wr_p,
                            const float* __restrict__ Wl_q, const float* __restrict__ Wr_q,
                            unsigned int* __restrict__ Wfrag) {
    int t = blockIdx.x * blockDim.x + threadIdx.x;
    if (t >= NKT_W * 16 * 32 * 8) return;
    const int v     = t & 7;
    const int lane  = (t >> 3) & 31;
    const int ntile = (t >> 8) & 15;
    const int kt    = t >> 12;
    const int n     = ntile * 16 + (lane & 15);
    const int khalf = lane >> 4;
    const int klo   = ((v & 4) << 2) + (khalf << 3) + ((v & 3) << 1);
    const int r     = kt * 32 + klo;
    const unsigned lo = f32_to_bf16_bits(wsrc_at(r,     n, Wl_p, Wr_p, Wl_q, Wr_q));
    const unsigned hi = f32_to_bf16_bits(wsrc_at(r + 1, n, Wl_p, Wr_p, Wl_q, Wr_q));
    Wfrag[t] = lo | (hi << 16);
}

// ---------------------------------------------------------------------------
// A fragment: per CDNA5 16-bit A 16x32 layout, lane's 16 elements are two
// contiguous 8-element runs: [k0+8*khalf .. +7] and [k0+16+8*khalf .. +7].
__device__ __forceinline__ v16bf load_a(const unsigned short* __restrict__ acrow, int k0) {
    const v8bf lo = *(const v8bf*)(acrow + k0);
    const v8bf hi = *(const v8bf*)(acrow + k0 + 16);
    return __builtin_shufflevector(lo, hi, 0, 1, 2, 3, 4, 5, 6, 7,
                                           8, 9, 10, 11, 12, 13, 14, 15);
}
__device__ __forceinline__ v16bf load_b(const unsigned int* __restrict__ Wfrag,
                                        int wtile, int ntile, int lane) {
    const v8i w = *(const v8i*)(Wfrag + (((wtile * 16 + ntile) * 32 + lane) << 3));
    return __builtin_bit_cast(v16bf, w);
}
__device__ __forceinline__ v8f wmma_bf16(v16bf a, v16bf b, v8f c) {
    return __builtin_amdgcn_wmma_f32_16x16x32_bf16(false, a, false, b,
                                                   (short)0, c, false, false);
}

// Fused GEMM: out = relu(0.5*(Acat_ext @ Wcat + b_p + b_q)), where Acat's
// x_lic block multiplies BOTH Wr_p (wtiles 4..11) and Wr_q (wtiles 18..25).
// 256 threads = 8 waves; wave = (mt 0..1, nt 0..3); each wave owns a
// 16(M) x 64(N) tile with 4 f32 accumulators.  M = 40000 = 1250*32 exactly.
__global__ __launch_bounds__(256)
void gemm_fused(const unsigned short* __restrict__ Acat,
                const unsigned int* __restrict__ Wfrag,
                const float* __restrict__ b_p, const float* __restrict__ b_q,
                float* __restrict__ out) {
    const int lane   = threadIdx.x & 31;
    const int wave   = threadIdx.x >> 5;
    const int mt     = wave >> 2;              // 0..1
    const int nt     = wave & 3;               // 0..3
    const int m0     = blockIdx.x * 32 + mt * 16;
    const int l15    = lane & 15;
    const int khalf  = lane >> 4;
    const int mrow   = m0 + l15;
    const int ntile0 = nt * 4;                 // 4 consecutive 16-col B tiles

    const unsigned short* acrow = Acat + (size_t)mrow * KA + khalf * 8;

    v8f acc0 = {}, acc1 = {}, acc2 = {}, acc3 = {};

    // Region 1: agg_p (A tiles 0..3 <-> W tiles 0..3, Wl_p)
#pragma unroll
    for (int kt = 0; kt < 4; ++kt) {
        const v16bf a = load_a(acrow, kt * 32);
        acc0 = wmma_bf16(a, load_b(Wfrag, kt, ntile0 + 0, lane), acc0);
        acc1 = wmma_bf16(a, load_b(Wfrag, kt, ntile0 + 1, lane), acc1);
        acc2 = wmma_bf16(a, load_b(Wfrag, kt, ntile0 + 2, lane), acc2);
        acc3 = wmma_bf16(a, load_b(Wfrag, kt, ntile0 + 3, lane), acc3);
    }
    // Region 2: x_lic (A tiles 4..11) -> Wr_p (W tiles 4..11) AND
    //                                    Wr_q (W tiles 18..25): 8 WMMAs per A.
#pragma unroll
    for (int kt = 4; kt < 12; ++kt) {
        const v16bf a = load_a(acrow, kt * 32);
        acc0 = wmma_bf16(a, load_b(Wfrag, kt, ntile0 + 0, lane), acc0);
        acc1 = wmma_bf16(a, load_b(Wfrag, kt, ntile0 + 1, lane), acc1);
        acc2 = wmma_bf16(a, load_b(Wfrag, kt, ntile0 + 2, lane), acc2);
        acc3 = wmma_bf16(a, load_b(Wfrag, kt, ntile0 + 3, lane), acc3);
        acc0 = wmma_bf16(a, load_b(Wfrag, kt + 14, ntile0 + 0, lane), acc0);
        acc1 = wmma_bf16(a, load_b(Wfrag, kt + 14, ntile0 + 1, lane), acc1);
        acc2 = wmma_bf16(a, load_b(Wfrag, kt + 14, ntile0 + 2, lane), acc2);
        acc3 = wmma_bf16(a, load_b(Wfrag, kt + 14, ntile0 + 3, lane), acc3);
    }
    // Region 3: agg_q (A tiles 12..17 <-> W tiles 12..17, Wl_q)
#pragma unroll
    for (int kt = 12; kt < 18; ++kt) {
        const v16bf a = load_a(acrow, kt * 32);
        acc0 = wmma_bf16(a, load_b(Wfrag, kt, ntile0 + 0, lane), acc0);
        acc1 = wmma_bf16(a, load_b(Wfrag, kt, ntile0 + 1, lane), acc1);
        acc2 = wmma_bf16(a, load_b(Wfrag, kt, ntile0 + 2, lane), acc2);
        acc3 = wmma_bf16(a, load_b(Wfrag, kt, ntile0 + 3, lane), acc3);
    }

    // Epilogue: 0.5*(acc + b_p + b_q), relu.  C/D layout: m = r + 8*khalf.
    const int ncol = ntile0 * 16 + l15;
    const float bb0 = b_p[ncol]      + b_q[ncol];
    const float bb1 = b_p[ncol + 16] + b_q[ncol + 16];
    const float bb2 = b_p[ncol + 32] + b_q[ncol + 32];
    const float bb3 = b_p[ncol + 48] + b_q[ncol + 48];
#pragma unroll
    for (int r = 0; r < 8; ++r) {
        const int m = m0 + khalf * 8 + r;
        float* o = out + (size_t)m * HDIM;
        o[ncol]      = fmaxf(0.5f * (acc0[r] + bb0), 0.0f);
        o[ncol + 16] = fmaxf(0.5f * (acc1[r] + bb1), 0.0f);
        o[ncol + 32] = fmaxf(0.5f * (acc2[r] + bb2), 0.0f);
        o[ncol + 48] = fmaxf(0.5f * (acc3[r] + bb3), 0.0f);
    }
}

// ---------------------------------------------------------------------------
extern "C" void kernel_launch(void* const* d_in, const int* in_sizes, int n_in,
                              void* d_out, int out_size, void* d_ws, size_t ws_size,
                              hipStream_t stream) {
    const float* x_lic = (const float*)d_in[0];
    const float* x_emp = (const float*)d_in[1];
    const float* x_con = (const float*)d_in[2];
    const float* Wl_p  = (const float*)d_in[3];
    const float* Wr_p  = (const float*)d_in[4];
    const float* b_p   = (const float*)d_in[5];
    const float* Wl_q  = (const float*)d_in[6];
    const float* Wr_q  = (const float*)d_in[7];
    const float* b_q   = (const float*)d_in[8];
    const int* ei_p_src = (const int*)d_in[9];
    const int* ei_p_dst = (const int*)d_in[10];
    const int* ei_q_src = (const int*)d_in[11];
    const int* ei_q_dst = (const int*)d_in[12];
    const int e_p = in_sizes[9];
    const int e_q = in_sizes[11];

    // Workspace layout:
    //   agg_p f32 [40000*128] | agg_q f32 [40000*192] | deg_p/deg_q f32 [40000 ea]
    //   | Acat bf16 [40000*576] | Wfrag u32 [26*16*32*8]   (~98.0 MB total)
    float* ws    = (float*)d_ws;
    float* agg_p = ws;
    float* agg_q = agg_p + (long)N_LIC * D_EMP;
    float* deg_p = agg_q + (long)N_LIC * D_CON;
    float* deg_q = deg_p + N_LIC;
    unsigned short* Acat  = (unsigned short*)(deg_q + N_LIC);
    unsigned int*   Wfrag = (unsigned int*)(Acat + (size_t)N_LIC * KA);

    const long nz = (long)N_LIC * D_EMP + (long)N_LIC * D_CON + 2L * N_LIC;
    zero_f32<<<(int)((nz + 255) / 256), 256, 0, stream>>>(ws, nz);

    pack_w_frag<<<(NKT_W * 16 * 32 * 8 + 255) / 256, 256, 0, stream>>>(
        Wl_p, Wr_p, Wl_q, Wr_q, Wfrag);

    scatter_p<<<(e_p + 7) / 8, 256, 0, stream>>>(x_emp, ei_p_src, ei_p_dst, agg_p, deg_p, e_p);
    scatter_q<<<(e_q + 7) / 8, 256, 0, stream>>>(x_con, ei_q_src, ei_q_dst, agg_q, deg_q, e_q);

    build_acat<<<N_LIC, KA, 0, stream>>>(x_lic, agg_p, agg_q, deg_p, deg_q, Acat);

    gemm_fused<<<N_LIC / 32, 256, 0, stream>>>(Acat, Wfrag, b_p, b_q, (float*)d_out);
}
